// ChaosAttentionBlock_29480655520348
// MI455X (gfx1250) — compile-verified, matrix operand order checked by reading, not run on hardware
//
#include <hip/hip_runtime.h>
#include <hip/hip_bf16.h>
#include <math.h>

// ---------------- constants (match reference) ----------------
#define DIMX     1024
#define HEADS    16
#define HEAD_DIM 64
#define FFD      2048
#define BATCH    2
#define SEQ      2048
#define ROWS     (BATCH*SEQ)   // 4096

typedef __attribute__((ext_vector_type(16))) _Float16 v16h;
typedef __attribute__((ext_vector_type(8)))  _Float16 h8;
typedef __attribute__((ext_vector_type(8)))  float    v8f;
typedef __attribute__((ext_vector_type(4)))  unsigned u32x4;
typedef __attribute__((ext_vector_type(8)))  unsigned u32x8;

// -------------------------------------------------------------
// Async global -> LDS (per-lane DMA, ASYNCcnt). GVS mode:
// mem = SADDR(64b) + VADDR(32b). vdst VGPR = LDS byte offset
// inside the block's allocation (hardware adds LDS_BASE).
// -------------------------------------------------------------
__device__ __forceinline__ void async_b128(unsigned ldsOff, unsigned gOff,
                                           const void* gBase) {
    asm volatile("global_load_async_to_lds_b128 %0, %1, %2 offset:0"
                 :: "v"(ldsOff), "v"(gOff), "s"(gBase)
                 : "memory");
}
__device__ __forceinline__ void wait_async0() {
    asm volatile("s_wait_asynccnt 0x0" ::: "memory");
}

// -------------------------------------------------------------
// Tensor Data Mover: 2D tile (64 x 128, f16) from a (rows x K)
// row-major tensor into LDS with hardware row padding:
// pad 4 DWORDs after every 32 DWORDs -> LDS row stride 72 halves.
// D# packing per CDNA5 ISA 8.3/8.4 (count=1, type=2=image,
// data_size=1 (2 bytes), pad_enable, pad_interval=4 (32 dw),
// pad_amount=3 (4 dw)). Tracked by TENSORcnt.
// -------------------------------------------------------------
__device__ __forceinline__ void tdm_load_tile_f16(unsigned ldsOff, const void* gTile,
                                                  unsigned K, unsigned rows) {
    unsigned long long ga = (unsigned long long)(size_t)gTile;
    u32x4 g0;
    g0[0] = 1u;                                   // count = 1 valid descriptor
    g0[1] = ldsOff;                               // lds_addr (bytes)
    g0[2] = (unsigned)ga;                         // global_addr[31:0]
    g0[3] = (unsigned)(ga >> 32) | (2u << 30);    // global_addr[56:32] | type=2
    u32x8 g1;
    g1[0] = (1u << 16)            // data_size = 2 bytes
          | (1u << 20)            // pad_enable
          | (4u << 22)            // pad_interval: 32 DWORDs
          | (3u << 25);           // pad_amount: 4 DWORDs
    g1[1] = (K & 0xFFFFu) << 16;                       // tensor_dim0[15:0]
    g1[2] = (K >> 16) | ((rows & 0xFFFFu) << 16);      // dim0[31:16] | dim1[15:0]
    g1[3] = (rows >> 16) | (64u << 16);                // dim1[31:16] | tile_dim0=64
    g1[4] = 128u;                                      // tile_dim1=128, tile_dim2=0
    g1[5] = K;                                         // tensor_dim0_stride[31:0]
    g1[6] = 0u;                                        // stride0[47:32] | stride1 lo
    g1[7] = 0u;
    asm volatile("tensor_load_to_lds %0, %1"
                 :: "s"(g0), "s"(g1)
                 : "memory");
}

// -------------------------------------------------------------
// Fragment load from LDS following CDNA5 16-bit A/B layout:
// lane = {m or n} = lane&15, khalf = lane>>4.
// VGPR0..3 hold K = khalf*8 + 0..7 ; VGPR4..7 hold K = 16 + khalf*8 + 0..7.
// -------------------------------------------------------------
__device__ __forceinline__ v16h frag16(const _Float16* rowPtr, int khalf) {
    h8 lo = *(const h8*)(rowPtr + khalf * 8);
    h8 hi = *(const h8*)(rowPtr + 16 + khalf * 8);
    v16h r;
#pragma unroll
    for (int i = 0; i < 8; ++i) { r[i] = lo[i]; r[i + 8] = hi[i]; }
    return r;
}

__device__ __forceinline__ v8f wmma_f16(v16h a, v16h b, v8f c) {
    return __builtin_amdgcn_wmma_f32_16x16x32_f16(
        /*neg_a=*/false, a, /*neg_b=*/false, b,
        /*c_mod=*/(short)0, c, /*reuse_a=*/false, /*reuse_b=*/false);
}

__device__ __forceinline__ v8f v8f_zero() {
    v8f z;
#pragma unroll
    for (int i = 0; i < 8; ++i) z[i] = 0.0f;
    return z;
}

// -------------------------------------------------------------
// fp32 -> f16 conversion (weights)
// -------------------------------------------------------------
__global__ __launch_bounds__(256)
void f32_to_f16_kernel(const float* __restrict__ in, _Float16* __restrict__ out, int n) {
    int i = blockIdx.x * 256 + threadIdx.x;
    if (i < n) out[i] = (_Float16)in[i];
}

// -------------------------------------------------------------
// RMSNorm: one block per row of 1024, f16 output
// -------------------------------------------------------------
__global__ __launch_bounds__(256)
void rmsnorm_f16_kernel(const float* __restrict__ X, const float* __restrict__ W,
                        _Float16* __restrict__ Out) {
    __shared__ float red[256];
    const int row = blockIdx.x;
    const float* xr = X + (size_t)row * DIMX;
    float v[4];
    float ss = 0.0f;
#pragma unroll
    for (int i = 0; i < 4; ++i) {
        v[i] = xr[threadIdx.x + i * 256];
        ss += v[i] * v[i];
    }
    red[threadIdx.x] = ss;
    __syncthreads();
    for (int off = 128; off > 0; off >>= 1) {
        if (threadIdx.x < off) red[threadIdx.x] += red[threadIdx.x + off];
        __syncthreads();
    }
    const float rs = rsqrtf(red[0] * (1.0f / DIMX) + 1e-6f);
#pragma unroll
    for (int i = 0; i < 4; ++i) {
        int c = threadIdx.x + i * 256;
        Out[(size_t)row * DIMX + c] = (_Float16)(v[i] * rs * W[c]);
    }
}

// -------------------------------------------------------------
// NT GEMM: C[M,N] = A[M,K](f16, row-major) * B[N,K](f16, row-major)^T
// 256 threads = 8 waves (2 M x 4 N); block tile 128x128.
// K staged in steps of 64 through double-buffered LDS filled by the
// Tensor Data Mover (wave 0 issues; TENSORcnt); 16 WMMAs per barrier.
// EPI: 0 = store f32; 1 = f32 residual add; 2 = SiLU -> f16
// -------------------------------------------------------------
#define BK   64
#define LDK  72            // 64 + 8 halves pad (matches TDM pad descriptor)
#define TILE_HALF (128 * LDK)          // halves per (A or B) tile
#define TILE_BYTE (TILE_HALF * 2)

template <int EPI>
__global__ __launch_bounds__(256)
void gemm_wmma_nt(const _Float16* __restrict__ A, const _Float16* __restrict__ Bw,
                  const float* __restrict__ Res, float* __restrict__ Cf,
                  _Float16* __restrict__ Ch, int M, int N, int K) {
    // single shared array so LDS byte offsets in the D# are exact:
    // buf b (0/1): A tile at (2b+0)*TILE, B tile at (2b+1)*TILE
    __shared__ __align__(16) _Float16 smem[4 * TILE_HALF];

    const int tid   = threadIdx.x;
    const int lane  = tid & 31;
    const int wave  = tid >> 5;
    const int wm    = wave >> 2;        // 0..1  -> 64 rows
    const int wn    = wave & 3;         // 0..3  -> 32 cols
    const int lr    = lane & 15;
    const int khalf = lane >> 4;
    const int mbase = blockIdx.y * 128;
    const int nbase = blockIdx.x * 128;

    v8f acc[4][2];
#pragma unroll
    for (int i = 0; i < 4; ++i)
#pragma unroll
        for (int j = 0; j < 2; ++j) acc[i][j] = v8f_zero();

    auto issue_tile = [&](int kk, int buf) {
        // wave 0 drives the DMA engine for the whole block
        tdm_load_tile_f16((unsigned)((buf * 2 + 0) * TILE_BYTE),
                          A + (size_t)mbase * K + kk, (unsigned)K, (unsigned)M);
        tdm_load_tile_f16((unsigned)((buf * 2 + 1) * TILE_BYTE),
                          Bw + (size_t)nbase * K + kk, (unsigned)K, (unsigned)N);
    };

    // prologue: fill buffer 0
    if (wave == 0) {
        issue_tile(0, 0);
        __builtin_amdgcn_s_wait_tensorcnt(0);
    }
    __syncthreads();

    int buf = 0;
    for (int kk = 0; kk < K; kk += BK) {
        if (wave == 0 && kk + BK < K) issue_tile(kk + BK, buf ^ 1);  // DMA next stage

        const _Float16* As = &smem[(buf * 2 + 0) * TILE_HALF];
        const _Float16* Bs = &smem[(buf * 2 + 1) * TILE_HALF];
#pragma unroll
        for (int ks = 0; ks < 2; ++ks) {                 // two 32-wide K sub-steps
            v16h af[4], bf[2];
#pragma unroll
            for (int mt = 0; mt < 4; ++mt)
                af[mt] = frag16(&As[(wm * 64 + mt * 16 + lr) * LDK + ks * 32], khalf);
#pragma unroll
            for (int nt = 0; nt < 2; ++nt)
                bf[nt] = frag16(&Bs[(wn * 32 + nt * 16 + lr) * LDK + ks * 32], khalf);
#pragma unroll
            for (int mt = 0; mt < 4; ++mt)
#pragma unroll
                for (int nt = 0; nt < 2; ++nt)
                    acc[mt][nt] = wmma_f16(af[mt], bf[nt], acc[mt][nt]);
        }

        if (wave == 0) __builtin_amdgcn_s_wait_tensorcnt(0);  // next stage landed
        __syncthreads();        // everyone done reading current stage
        buf ^= 1;
    }

    // epilogue; C fragment: row = 8*khalf + j, col = lr within each 16x16 tile
#pragma unroll
    for (int mt = 0; mt < 4; ++mt) {
#pragma unroll
        for (int nt = 0; nt < 2; ++nt) {
#pragma unroll
            for (int j = 0; j < 8; ++j) {
                int r = mbase + wm * 64 + mt * 16 + 8 * khalf + j;
                int c = nbase + wn * 32 + nt * 16 + lr;
                float val = acc[mt][nt][j];
                if (EPI == 0) {
                    Cf[(size_t)r * N + c] = val;
                } else if (EPI == 1) {
                    Cf[(size_t)r * N + c] = Res[(size_t)r * N + c] + val;
                } else {  // SiLU -> f16
                    float s = val / (1.0f + __expf(-val));
                    Ch[(size_t)r * N + c] = (_Float16)s;
                }
            }
        }
    }
}

// -------------------------------------------------------------
// RoPE + reorder into per-head (B,H,S,Dh) f16 layout.
// Q is pre-scaled by 1/sqrt(HEAD_DIM) so attention skips the mul.
// -------------------------------------------------------------
__global__ __launch_bounds__(256)
void rope_split_kernel(const float* __restrict__ QKV, _Float16* __restrict__ Qh,
                       _Float16* __restrict__ Kh, _Float16* __restrict__ Vh) {
    const int row = blockIdx.x;          // 0..ROWS-1
    const int b = row / SEQ, s = row % SEQ;
    const float* base = QKV + (size_t)row * (3 * DIMX);
    const int t = threadIdx.x;
    const float qscale = 0.125f;         // 1/sqrt(64)
#pragma unroll
    for (int u0 = 0; u0 < 2; ++u0) {
        int u = t + u0 * 256;            // 0..511  = (head, pair)
        int h = u >> 5, p = u & 31;
        float inv = __expf(-(2.0f * (float)p / (float)HEAD_DIM) * 9.210340372f); // ln(1e4)
        float ang = (float)s * inv;
        float sn, cs;
        __sincosf(ang, &sn, &cs);
        size_t dst = ((size_t)(b * HEADS + h) * SEQ + s) * HEAD_DIM + 2 * p;
        float qe = base[h * HEAD_DIM + 2 * p];
        float qo = base[h * HEAD_DIM + 2 * p + 1];
        Qh[dst]     = (_Float16)((qe * cs - qo * sn) * qscale);
        Qh[dst + 1] = (_Float16)((qo * cs + qe * sn) * qscale);
        float ke = base[DIMX + h * HEAD_DIM + 2 * p];
        float ko = base[DIMX + h * HEAD_DIM + 2 * p + 1];
        Kh[dst]     = (_Float16)(ke * cs - ko * sn);
        Kh[dst + 1] = (_Float16)(ko * cs + ke * sn);
    }
#pragma unroll
    for (int i = 0; i < 4; ++i) {
        int d = t + i * 256;             // 0..1023
        int h = d >> 6, dd = d & 63;
        Vh[((size_t)(b * HEADS + h) * SEQ + s) * HEAD_DIM + dd] =
            (_Float16)base[2 * DIMX + d];
    }
}

// -------------------------------------------------------------
// Flash attention (causal). Block = 128 threads (4 waves).
// blockIdx.x = b*HEADS+h, blockIdx.y = 64-row Q tile.
// Q/K tiles arrive via async global->LDS DMA; V is transposed by hand.
// Single loop body (runtime diag flag) keeps VGPR use <= 256.
// -------------------------------------------------------------
#define ATT_LD  72         // padded row stride in halves
#define ATT_TILE (64 * ATT_LD)

__global__ __launch_bounds__(128)
void attn_fwd_kernel(const _Float16* __restrict__ Qh, const _Float16* __restrict__ Kh,
                     const _Float16* __restrict__ Vh, _Float16* __restrict__ Oh) {
    // single shared array: Qs | Ks | Vt | Ps
    __shared__ __align__(16) _Float16 sm[4 * ATT_TILE];
    _Float16* Qs = sm;
    _Float16* Ks = sm + ATT_TILE;
    _Float16* Vt = sm + 2 * ATT_TILE;   // V transposed: [d][key]
    _Float16* Ps = sm + 3 * ATT_TILE;

    const int tid   = threadIdx.x;
    const int lane  = tid & 31;
    const int wave  = tid >> 5;
    const int lr    = lane & 15;
    const int khalf = lane >> 4;
    const int bh    = blockIdx.x;
    const int qt    = blockIdx.y;
    const int b     = bh / HEADS, h = bh % HEADS;
    const size_t headBase = (size_t)bh * SEQ * HEAD_DIM;
    const _Float16* Qbase = Qh + headBase;
    const _Float16* Kbase = Kh + headBase;
    const _Float16* Vbase = Vh + headBase;
    const int qbase = qt * 64;

    // per-thread chunk coords: 64x64 f16 tile = 512 x 16B chunks, 8 per row
    int crow[4], cco[4];
#pragma unroll
    for (int c = 0; c < 4; ++c) {
        int chunk = tid + c * 128;       // 0..511
        crow[c] = chunk >> 3;            // 0..63
        cco[c]  = (chunk & 7) * 8;       // 0..56 halves
    }

    // async load Q tile
#pragma unroll
    for (int c = 0; c < 4; ++c) {
        unsigned g = (unsigned)((((qbase + crow[c]) * HEAD_DIM) + cco[c]) * 2);
        unsigned l = (unsigned)((crow[c] * ATT_LD + cco[c]) * 2);   // Qs at offset 0
        async_b128(l, g, Qbase);
    }
    wait_async0();
    __syncthreads();

    v16h qf0 = frag16(&Qs[(wave * 16 + lr) * ATT_LD + 0], khalf);
    v16h qf1 = frag16(&Qs[(wave * 16 + lr) * ATT_LD + 32], khalf);

    v8f o[4];
#pragma unroll
    for (int i = 0; i < 4; ++i) o[i] = v8f_zero();
    float rM[8], rL[8];
#pragma unroll
    for (int j = 0; j < 8; ++j) { rM[j] = -1e30f; rL[j] = 0.0f; }

    for (int kt = 0; kt <= qt; ++kt) {
        const int kb = kt * 64;
        const bool diag = (kt == qt);
        // async K tile ([key][d] row-major)
#pragma unroll
        for (int c = 0; c < 4; ++c) {
            unsigned g = (unsigned)((((kb + crow[c]) * HEAD_DIM) + cco[c]) * 2);
            unsigned l = (unsigned)((ATT_TILE + crow[c] * ATT_LD + cco[c]) * 2);
            async_b128(l, g, Kbase);
        }
        // V tile transposed by hand ([d][key])
#pragma unroll
        for (int e = 0; e < 32; ++e) {
            int idx = tid + e * 128;     // 0..4095
            int key = idx >> 6, d = idx & 63;
            Vt[d * ATT_LD + key] = Vbase[(size_t)(kb + key) * HEAD_DIM + d];
        }
        wait_async0();
        __syncthreads();

        // scores S = Q K^T (Q already carries 1/sqrt(Dh))
        v8f sfr[4];
#pragma unroll
        for (int nt = 0; nt < 4; ++nt) {
            v8f c0 = v8f_zero();
            v16h b0 = frag16(&Ks[(nt * 16 + lr) * ATT_LD + 0], khalf);
            v16h b1 = frag16(&Ks[(nt * 16 + lr) * ATT_LD + 32], khalf);
            c0 = wmma_f16(qf0, b0, c0);
            c0 = wmma_f16(qf1, b1, c0);
            sfr[nt] = c0;
        }

        if (diag) {
#pragma unroll
            for (int nt = 0; nt < 4; ++nt) {
                int col = kb + nt * 16 + lr;
#pragma unroll
                for (int j = 0; j < 8; ++j) {
                    int row = qbase + wave * 16 + 8 * khalf + j;
                    if (col > row) sfr[nt][j] = -1e30f;
                }
            }
        }

        // online softmax per row (rows live in lane halves; reduce over 16 lanes)
#pragma unroll
        for (int j = 0; j < 8; ++j) {
            float m = fmaxf(fmaxf(sfr[0][j], sfr[1][j]), fmaxf(sfr[2][j], sfr[3][j]));
            for (int off = 1; off < 16; off <<= 1)
                m = fmaxf(m, __shfl_xor(m, off, 32));
            float newM = fmaxf(rM[j], m);
            float alpha = __expf(rM[j] - newM);
            float sum = 0.0f;
#pragma unroll
            for (int nt = 0; nt < 4; ++nt) {
                float p = __expf(sfr[nt][j] - newM);
                sfr[nt][j] = p;
                sum += p;
            }
            for (int off = 1; off < 16; off <<= 1)
                sum += __shfl_xor(sum, off, 32);
            rL[j] = rL[j] * alpha + sum;
            rM[j] = newM;
#pragma unroll
            for (int dt = 0; dt < 4; ++dt) o[dt][j] *= alpha;
        }

        // P -> LDS (each wave writes/reads only its own 16 rows; per-wave DS order)
#pragma unroll
        for (int nt = 0; nt < 4; ++nt)
#pragma unroll
            for (int j = 0; j < 8; ++j)
                Ps[(wave * 16 + 8 * khalf + j) * ATT_LD + nt * 16 + lr] =
                    (_Float16)sfr[nt][j];

        v16h pf0 = frag16(&Ps[(wave * 16 + lr) * ATT_LD + 0], khalf);
        v16h pf1 = frag16(&Ps[(wave * 16 + lr) * ATT_LD + 32], khalf);
#pragma unroll
        for (int dt = 0; dt < 4; ++dt) {
            v16h v0 = frag16(&Vt[(dt * 16 + lr) * ATT_LD + 0], khalf);
            v16h v1 = frag16(&Vt[(dt * 16 + lr) * ATT_LD + 32], khalf);
            o[dt] = wmma_f16(pf0, v0, o[dt]);
            o[dt] = wmma_f16(pf1, v1, o[dt]);
        }
        __syncthreads();
    }

    // normalize and store into (B*S, DIM) f16 layout for out-proj GEMM
#pragma unroll
    for (int dt = 0; dt < 4; ++dt)
#pragma unroll
        for (int j = 0; j < 8; ++j) {
            int row = qbase + wave * 16 + 8 * khalf + j;
            float val = o[dt][j] / rL[j];
            Oh[((size_t)(b * SEQ + row)) * DIMX + h * HEAD_DIM + dt * 16 + lr] =
                (_Float16)val;
        }
}

// -------------------------------------------------------------
// host-side orchestration
// -------------------------------------------------------------
extern "C" void kernel_launch(void* const* d_in, const int* in_sizes, int n_in,
                              void* d_out, int out_size, void* d_ws, size_t ws_size,
                              hipStream_t stream) {
    const float* x      = (const float*)d_in[0];
    const float* w_in_n = (const float*)d_in[1];
    const float* w_ff_n = (const float*)d_in[2];
    const float* w_qkv  = (const float*)d_in[3];
    const float* w_out  = (const float*)d_in[4];
    const float* w_fc   = (const float*)d_in[5];
    const float* w_proj = (const float*)d_in[6];
    float* out = (float*)d_out;

    size_t off = 0;
    auto wsAlloc = [&](size_t bytes) -> void* {
        void* p = (char*)d_ws + off;
        off += (bytes + 255) & ~(size_t)255;
        return p;
    };

    _Float16* wqkv_h  = (_Float16*)wsAlloc((size_t)3 * DIMX * DIMX * 2);
    _Float16* wout_h  = (_Float16*)wsAlloc((size_t)DIMX * DIMX * 2);
    _Float16* wfc_h   = (_Float16*)wsAlloc((size_t)FFD * DIMX * 2);
    _Float16* wproj_h = (_Float16*)wsAlloc((size_t)DIMX * FFD * 2);
    _Float16* hN      = (_Float16*)wsAlloc((size_t)ROWS * DIMX * 2);
    float*    qkv     = (float*)   wsAlloc((size_t)ROWS * 3 * DIMX * 4);
    _Float16* qh      = (_Float16*)wsAlloc((size_t)ROWS * DIMX * 2);
    _Float16* kh      = (_Float16*)wsAlloc((size_t)ROWS * DIMX * 2);
    _Float16* vh      = (_Float16*)wsAlloc((size_t)ROWS * DIMX * 2);
    _Float16* attnb   = (_Float16*)wsAlloc((size_t)ROWS * DIMX * 2);
    float*    x1      = (float*)   wsAlloc((size_t)ROWS * DIMX * 4);
    _Float16* h2      = (_Float16*)wsAlloc((size_t)ROWS * DIMX * 2);
    _Float16* hid     = (_Float16*)wsAlloc((size_t)ROWS * FFD * 2);

    // 1) weight fp32 -> f16
    {
        int n;
        n = 3 * DIMX * DIMX;
        f32_to_f16_kernel<<<(n + 255) / 256, 256, 0, stream>>>(w_qkv, wqkv_h, n);
        n = DIMX * DIMX;
        f32_to_f16_kernel<<<(n + 255) / 256, 256, 0, stream>>>(w_out, wout_h, n);
        n = FFD * DIMX;
        f32_to_f16_kernel<<<(n + 255) / 256, 256, 0, stream>>>(w_fc, wfc_h, n);
        n = DIMX * FFD;
        f32_to_f16_kernel<<<(n + 255) / 256, 256, 0, stream>>>(w_proj, wproj_h, n);
    }

    // 2) pre-attention RMSNorm
    rmsnorm_f16_kernel<<<ROWS, 256, 0, stream>>>(x, w_in_n, hN);

    // 3) QKV GEMM: (4096 x 3072) = hN (4096x1024) * w_qkv^T
    {
        dim3 g(3 * DIMX / 128, ROWS / 128);
        gemm_wmma_nt<0><<<g, 256, 0, stream>>>(hN, wqkv_h, nullptr, qkv, nullptr,
                                               ROWS, 3 * DIMX, DIMX);
    }

    // 4) RoPE + split to per-head layout (Q pre-scaled)
    rope_split_kernel<<<ROWS, 256, 0, stream>>>(qkv, qh, kh, vh);

    // 5) causal flash attention
    {
        dim3 g(BATCH * HEADS, SEQ / 64);
        attn_fwd_kernel<<<g, 128, 0, stream>>>(qh, kh, vh, attnb);
    }

    // 6) out projection + residual: x1 = x + attn * w_out^T
    {
        dim3 g(DIMX / 128, ROWS / 128);
        gemm_wmma_nt<1><<<g, 256, 0, stream>>>(attnb, wout_h, x, x1, nullptr,
                                               ROWS, DIMX, DIMX);
    }

    // 7) FF RMSNorm
    rmsnorm_f16_kernel<<<ROWS, 256, 0, stream>>>(x1, w_ff_n, h2);

    // 8) FC GEMM + SiLU -> f16 hidden
    {
        dim3 g(FFD / 128, ROWS / 128);
        gemm_wmma_nt<2><<<g, 256, 0, stream>>>(h2, wfc_h, nullptr, nullptr, hid,
                                               ROWS, FFD, DIMX);
    }

    // 9) proj GEMM + residual -> final output
    {
        dim3 g(DIMX / 128, ROWS / 128);
        gemm_wmma_nt<1><<<g, 256, 0, stream>>>(hid, wproj_h, x1, out, nullptr,
                                               ROWS, DIMX, FFD);
    }
}
